// AlphaStarPolicyNetwork_59785944761265
// MI455X (gfx1250) — matrix operand order repaired and checked by reading, not compiled, not run on hardware
//
#include <hip/hip_runtime.h>
#include <cstddef>
#include <cstdint>

// ---------------- problem constants ----------------
#define Hdim   1024
#define NHEADS 16
#define HDh    64
#define Tseq   7
#define Bsz    8192
#define MT     (Bsz * Tseq)      // 57344 transformer rows
#define FFN    4096

typedef __attribute__((ext_vector_type(16))) __bf16 v16bf;
typedef __attribute__((ext_vector_type(8)))  __bf16 v8bf;
typedef __attribute__((ext_vector_type(8)))  float  v8f;

union FragU { v16bf v; v8bf h[2]; };

// A-matrix 16x32 bf16 fragment: lanes 0-15 hold K=0..7 (vgpr0-3) and K=16..23
// (vgpr4-7); lanes 16-31 hold K=8..15 and K=24..31  (CDNA5 ISA 7.12.2).
__device__ __forceinline__ v16bf load_frag_a(const __bf16* p) {
  FragU f;
  f.h[0] = *(const v8bf*)(p);        // contiguous 8 bf16 (b128)
  f.h[1] = *(const v8bf*)(p + 16);   // +32B: second K block
  return f.v;
}
// B-matrix 32x16 fragment (rows of W^T / LDS tile): lanes 0-15 hold K=0..7 and
// K=16..23; lanes 16-31 hold K=8..15 and K=24..31 (2x ds_load_b128).
__device__ __forceinline__ v16bf load_frag_b(const __bf16* p) {
  FragU f;
  f.h[0] = *(const v8bf*)(p);
  f.h[1] = *(const v8bf*)(p + 16);
  return f.v;
}

// gfx1250 async copy global -> LDS (tracked with ASYNCcnt), 16B per lane.
__device__ __forceinline__ void async_load_b128(const __bf16* lds_dst,
                                                const __bf16* gsrc) {
  unsigned l = (unsigned)(uintptr_t)lds_dst;           // LDS byte offset
  unsigned long long g = (unsigned long long)(uintptr_t)gsrc;
  asm volatile("global_load_async_to_lds_b128 %0, %1, off"
               :: "v"(l), "v"(g) : "memory");
}

__device__ __forceinline__ float gelu_erf(float v) {
  return 0.5f * v * (1.0f + erff(v * 0.70710678118654752f));
}
__device__ __forceinline__ float sigmoidf_(float v) {
  return 1.0f / (1.0f + __expf(-v));
}

enum { EPI_F32 = 0, EPI_RES = 1, EPI_BF16 = 2, EPI_GELU_BF16 = 3 };

// ---------------------------------------------------------------------------
// WMMA GEMM:  out[M,N] = A[M,K](bf16) * Wt[N,K]^T(bf16) + bias, fused epilogue
// Block: 256 thr = 8 waves covering 256 x 64; each wave owns 32 x 64
// (2 A-frags x 4 B-frags = 8 WMMAs per K-step). B tile (64x32 bf16 = 4KB) is
// async-staged into LDS, double buffered, shared by all 8 waves.
// ---------------------------------------------------------------------------
template<int EPI>
__global__ __launch_bounds__(256)
void gemm_bf16(const __bf16* __restrict__ A, const __bf16* __restrict__ Wt,
               const float* __restrict__ bias, const float* resid,
               float* outF, __bf16* outB, int M, int N, int K)
{
  __shared__ __bf16 Bs[2][64][32];                     // 8 KB double buffer

  const int tid  = threadIdx.x;
  const int lane = tid & 31;
  const int wave = tid >> 5;
  const int half = lane >> 4;
  const int l16  = lane & 15;
  const int m0 = blockIdx.y * 256 + wave * 32;
  const int n0 = blockIdx.x * 64;

  // async staging assignment: thread -> (tile row, 8-elem segment)
  const int srow = tid >> 2;                           // 0..63
  const int sseg = (tid & 3) * 8;                      // 0,8,16,24
  const __bf16* bgl = Wt + (size_t)(n0 + srow) * K + sseg;

  v8f acc[2][4] = {};
  const __bf16* arow0 = A + (size_t)(m0 + l16) * K + half * 8;
  const __bf16* arow1 = arow0 + (size_t)16 * K;

  // prologue: fill buffer 0
  async_load_b128(&Bs[0][srow][sseg], bgl);

  const int nsteps = K / 32;
  for (int s = 0; s < nsteps; ++s) {
    const int cur = s & 1;
    if (s + 1 < nsteps) {
      async_load_b128(&Bs[cur ^ 1][srow][sseg], bgl + (size_t)(s + 1) * 32);
      asm volatile("s_wait_asynccnt 1" ::: "memory");  // cur buffer landed
    } else {
      asm volatile("s_wait_asynccnt 0" ::: "memory");
    }
    __syncthreads();                                   // tile visible to WGP

    const int k = s * 32;
    __builtin_prefetch(arow0 + k + 256, 0, 1);         // global_prefetch
    v16bf a0 = load_frag_a(arow0 + k);
    v16bf a1 = load_frag_a(arow1 + k);
#pragma unroll
    for (int j = 0; j < 4; ++j) {
      v16bf bf = load_frag_b(&Bs[cur][j * 16 + l16][half * 8]);
      acc[0][j] = __builtin_amdgcn_wmma_f32_16x16x32_bf16(
          false, a0, false, bf, (short)0, acc[0][j], false, false);
      acc[1][j] = __builtin_amdgcn_wmma_f32_16x16x32_bf16(
          false, a1, false, bf, (short)0, acc[1][j], false, false);
    }
    __syncthreads();                                   // done reading cur buf
  }

  // C/D layout: vgpr r -> row subtile_m0 + half*8 + r ; col n0 + j*16 + l16
#pragma unroll
  for (int mi = 0; mi < 2; ++mi) {
#pragma unroll
    for (int j = 0; j < 4; ++j) {
      const int col = n0 + j * 16 + l16;
      const float bv = bias ? bias[col] : 0.0f;
#pragma unroll
      for (int r = 0; r < 8; ++r) {
        const int row = m0 + mi * 16 + half * 8 + r;
        const size_t idx = (size_t)row * N + col;
        float v = acc[mi][j][r] + bv;
        if (EPI == EPI_F32)       outF[idx] = v;
        if (EPI == EPI_RES)       outF[idx] = v + resid[idx];
        if (EPI == EPI_BF16)      outB[idx] = (__bf16)v;
        if (EPI == EPI_GELU_BF16) outB[idx] = (__bf16)gelu_erf(v);
      }
    }
  }
}

// ---------------------------------------------------------------------------
// Weight convert + transpose:  Wt[n*K + k] = bf16(W[k*N + n])
// ---------------------------------------------------------------------------
__global__ __launch_bounds__(256)
void convert_wt(const float* __restrict__ W, __bf16* __restrict__ Wt,
                int K, int N)
{
  size_t i = (size_t)blockIdx.x * 256 + threadIdx.x;
  if (i >= (size_t)K * N) return;
  int n = (int)(i / K);
  int k = (int)(i % K);
  Wt[i] = (__bf16)W[(size_t)k * N + n];
}

// ---------------------------------------------------------------------------
// Entity encoder: one block per batch row; e-th entity group.
// out = gelu(LN(inp @ W + b)) + type_embed[e]
// ---------------------------------------------------------------------------
__global__ __launch_bounds__(256)
void entity_encoder(const float* __restrict__ state, int off, int dim, int e,
                    const float* __restrict__ W, const float* __restrict__ b,
                    const float* __restrict__ g, const float* __restrict__ be,
                    const float* __restrict__ te, float* __restrict__ x)
{
  const int bi = blockIdx.x;
  const float* inp = state + (size_t)bi * 22 + off;
  float in[4];
  for (int d = 0; d < dim; ++d) in[d] = inp[d];

  float vals[4], s = 0.f, ss = 0.f;
#pragma unroll
  for (int i = 0; i < 4; ++i) {
    int h = threadIdx.x + i * 256;
    float a = b[h];
    for (int d = 0; d < dim; ++d) a += in[d] * W[d * Hdim + h];
    vals[i] = a; s += a; ss += a * a;
  }
  __shared__ float r0[256], r1[256];
  r0[threadIdx.x] = s; r1[threadIdx.x] = ss;
  __syncthreads();
  for (int o = 128; o > 0; o >>= 1) {
    if (threadIdx.x < o) { r0[threadIdx.x] += r0[threadIdx.x + o];
                           r1[threadIdx.x] += r1[threadIdx.x + o]; }
    __syncthreads();
  }
  const float mean = r0[0] * (1.0f / Hdim);
  const float var  = r1[0] * (1.0f / Hdim) - mean * mean;
  const float rstd = rsqrtf(var + 1e-5f);
  float* out = x + ((size_t)bi * Tseq + e) * Hdim;
#pragma unroll
  for (int i = 0; i < 4; ++i) {
    int h = threadIdx.x + i * 256;
    float ln = (vals[i] - mean) * rstd * g[h] + be[h];
    out[h] = gelu_erf(ln) + te[h];
  }
}

// ---------------------------------------------------------------------------
// LayerNorm over H=1024, bf16 output (GEMM A operand).
// ---------------------------------------------------------------------------
__global__ __launch_bounds__(256)
void ln_rows(const float* __restrict__ x, const float* __restrict__ g,
             const float* __restrict__ b, __bf16* __restrict__ out)
{
  const size_t base = (size_t)blockIdx.x * Hdim;
  float vals[4], s = 0.f, ss = 0.f;
#pragma unroll
  for (int i = 0; i < 4; ++i) {
    int c = threadIdx.x + i * 256;
    float v = x[base + c];
    vals[i] = v; s += v; ss += v * v;
  }
  __shared__ float r0[256], r1[256];
  r0[threadIdx.x] = s; r1[threadIdx.x] = ss;
  __syncthreads();
  for (int o = 128; o > 0; o >>= 1) {
    if (threadIdx.x < o) { r0[threadIdx.x] += r0[threadIdx.x + o];
                           r1[threadIdx.x] += r1[threadIdx.x + o]; }
    __syncthreads();
  }
  const float mean = r0[0] * (1.0f / Hdim);
  const float var  = r1[0] * (1.0f / Hdim) - mean * mean;
  const float rstd = rsqrtf(var + 1e-5f);
#pragma unroll
  for (int i = 0; i < 4; ++i) {
    int c = threadIdx.x + i * 256;
    out[base + c] = (__bf16)((vals[i] - mean) * rstd * g[c] + b[c]);
  }
}

// ---------------------------------------------------------------------------
// Fused RoPE + attention for one (batch, head): T=7, HD=64. Output as bf16.
// ---------------------------------------------------------------------------
__global__ __launch_bounds__(64)
void attn_rope(const float* __restrict__ q, const float* __restrict__ k,
               const float* __restrict__ v, __bf16* __restrict__ outb)
{
  const int bh = blockIdx.x;
  const int b  = bh >> 4;          // / NHEADS
  const int hh = bh & 15;
  const int d  = threadIdx.x;      // 0..63
  __shared__ float Q[Tseq][HDh], K[Tseq][HDh], V[Tseq][HDh], S[Tseq][8];

  const size_t base = (size_t)b * Tseq * Hdim + hh * HDh + d;
  const float inv_freq = __powf(10000.0f, -(float)(d & 31) / 32.0f);

#pragma unroll
  for (int t = 0; t < Tseq; ++t) {
    Q[t][d] = q[base + (size_t)t * Hdim];
    K[t][d] = k[base + (size_t)t * Hdim];
    V[t][d] = v[base + (size_t)t * Hdim];
  }
  __syncthreads();

  float qn[Tseq], kn[Tseq];
#pragma unroll
  for (int t = 0; t < Tseq; ++t) {
    const float ang = (float)t * inv_freq;
    const float c = __cosf(ang), s = __sinf(ang);
    const float qr = (d < 32) ? -Q[t][d + 32] : Q[t][d - 32];
    const float kr = (d < 32) ? -K[t][d + 32] : K[t][d - 32];
    qn[t] = Q[t][d] * c + qr * s;
    kn[t] = K[t][d] * c + kr * s;
  }
  __syncthreads();
#pragma unroll
  for (int t = 0; t < Tseq; ++t) { Q[t][d] = qn[t]; K[t][d] = kn[t]; }
  __syncthreads();

  if (d < Tseq * Tseq) {
    const int i = d / Tseq, j = d % Tseq;
    float s = 0.f;
    for (int dd = 0; dd < HDh; ++dd) s += Q[i][dd] * K[j][dd];
    S[i][j] = s * 0.125f;            // 1/sqrt(64)
  }
  __syncthreads();
  if (d < Tseq) {
    float mx = S[d][0];
    for (int j = 1; j < Tseq; ++j) mx = fmaxf(mx, S[d][j]);
    float e[Tseq], sum = 0.f;
    for (int j = 0; j < Tseq; ++j) { e[j] = __expf(S[d][j] - mx); sum += e[j]; }
    const float is = 1.0f / sum;
    for (int j = 0; j < Tseq; ++j) S[d][j] = e[j] * is;
  }
  __syncthreads();
#pragma unroll
  for (int t = 0; t < Tseq; ++t) {
    float o = 0.f;
    for (int j = 0; j < Tseq; ++j) o += S[t][j] * V[j][d];
    outb[base + (size_t)t * Hdim] = (__bf16)o;
  }
}

// ---------------------------------------------------------------------------
// Mean over T=7 entities -> xm fp32 + bf16 copy (GEMM A operand).
// ---------------------------------------------------------------------------
__global__ __launch_bounds__(256)
void mean_t(const float* __restrict__ x, float* __restrict__ xm,
            __bf16* __restrict__ ab)
{
  size_t i = (size_t)blockIdx.x * 256 + threadIdx.x;
  if (i >= (size_t)Bsz * Hdim) return;
  const int b = (int)(i >> 10), c = (int)(i & 1023);
  float s = 0.f;
#pragma unroll
  for (int t = 0; t < Tseq; ++t) s += x[((size_t)b * Tseq + t) * Hdim + c];
  const float m = s * (1.0f / 7.0f);
  xm[i] = m;
  ab[i] = (__bf16)m;
}

// ---------------------------------------------------------------------------
// Gated block combine:  x = LN(x + sigmoid(gpre) * h)  (fp32 + bf16 copy)
// ---------------------------------------------------------------------------
__global__ __launch_bounds__(256)
void gated_combine(float* __restrict__ x, const float* __restrict__ gpre,
                   const float* __restrict__ h, const float* __restrict__ gamma,
                   const float* __restrict__ beta, __bf16* __restrict__ ab)
{
  const size_t base = (size_t)blockIdx.x * Hdim;
  float vals[4], s = 0.f, ss = 0.f;
#pragma unroll
  for (int i = 0; i < 4; ++i) {
    int c = threadIdx.x + i * 256;
    float u = x[base + c] + sigmoidf_(gpre[base + c]) * h[base + c];
    vals[i] = u; s += u; ss += u * u;
  }
  __shared__ float r0[256], r1[256];
  r0[threadIdx.x] = s; r1[threadIdx.x] = ss;
  __syncthreads();
  for (int o = 128; o > 0; o >>= 1) {
    if (threadIdx.x < o) { r0[threadIdx.x] += r0[threadIdx.x + o];
                           r1[threadIdx.x] += r1[threadIdx.x + o]; }
    __syncthreads();
  }
  const float mean = r0[0] * (1.0f / Hdim);
  const float var  = r1[0] * (1.0f / Hdim) - mean * mean;
  const float rstd = rsqrtf(var + 1e-5f);
#pragma unroll
  for (int i = 0; i < 4; ++i) {
    int c = threadIdx.x + i * 256;
    float o = (vals[i] - mean) * rstd * gamma[c] + beta[c];
    x[base + c] = o;
    ab[base + c] = (__bf16)o;
  }
}

// ---------------------------------------------------------------------------
// Final head: trade[2]/atype[3] from 512-wide gelu features, then 5x4 logits.
// ---------------------------------------------------------------------------
__global__ __launch_bounds__(256)
void head_final(const __bf16* __restrict__ t1, const __bf16* __restrict__ a1,
                const float* __restrict__ Wt2, const float* __restrict__ bt2,
                const float* __restrict__ Wa2, const float* __restrict__ ba2,
                const float* __restrict__ Wc,  const float* __restrict__ bc,
                float* __restrict__ out)
{
  const int b = blockIdx.x * 256 + threadIdx.x;
  if (b >= Bsz) return;
  float tr0 = bt2[0], tr1 = bt2[1];
  for (int i = 0; i < 512; ++i) {
    float v = (float)t1[(size_t)b * 512 + i];
    tr0 += v * Wt2[i * 2 + 0];
    tr1 += v * Wt2[i * 2 + 1];
  }
  float a0 = ba2[0], a1v = ba2[1], a2 = ba2[2];
  for (int i = 0; i < 512; ++i) {
    float v = (float)a1[(size_t)b * 512 + i];
    a0  += v * Wa2[i * 3 + 0];
    a1v += v * Wa2[i * 3 + 1];
    a2  += v * Wa2[i * 3 + 2];
  }
  float comb[5] = {tr0, tr1, a0, a1v, a2};
#pragma unroll
  for (int o = 0; o < 4; ++o) {
    float s = bc[o];
    for (int i = 0; i < 5; ++i) s += comb[i] * Wc[i * 4 + o];
    out[(size_t)b * 4 + o] = s;
  }
}

// ---------------------------------------------------------------------------
// host orchestration
// ---------------------------------------------------------------------------
static void launch_gemm(int epi, const __bf16* A, const __bf16* Wt,
                        const float* bias, const float* resid,
                        float* oF, __bf16* oB, int M, int N, int K,
                        hipStream_t s)
{
  dim3 g(N / 64, M / 256), b(256);
  switch (epi) {
  case EPI_F32:       gemm_bf16<EPI_F32>      <<<g, b, 0, s>>>(A, Wt, bias, resid, oF, oB, M, N, K); break;
  case EPI_RES:       gemm_bf16<EPI_RES>      <<<g, b, 0, s>>>(A, Wt, bias, resid, oF, oB, M, N, K); break;
  case EPI_BF16:      gemm_bf16<EPI_BF16>     <<<g, b, 0, s>>>(A, Wt, bias, resid, oF, oB, M, N, K); break;
  case EPI_GELU_BF16: gemm_bf16<EPI_GELU_BF16><<<g, b, 0, s>>>(A, Wt, bias, resid, oF, oB, M, N, K); break;
  }
}

extern "C" void kernel_launch(void* const* d_in, const int* in_sizes, int n_in,
                              void* d_out, int out_size, void* d_ws, size_t ws_size,
                              hipStream_t stream)
{
  (void)in_sizes; (void)n_in; (void)out_size; (void)ws_size;
  auto F = [&](int i) { return (const float*)d_in[i]; };

  // ---- input map (setup_inputs dict order, depth-first) ----
  int idx = 0;
  const float* state = F(idx++);                               // [8192,22]
  const float *entW[7], *entB[7], *entG[7], *entBe[7];
  for (int e = 0; e < 7; ++e) {
    entW[e] = F(idx++); entB[e] = F(idx++);
    entG[e] = F(idx++); entBe[e] = F(idx++);
  }
  const float* type_embed = F(idx++);                          // [7,1024]
  struct Layer { const float *Wq,*bq,*Wk,*bk,*Wv,*bv,*Wo,*bo,
                             *W1,*b1,*W2,*b2,*n1g,*n1b,*n2g,*n2b; } L[4];
  for (int l = 0; l < 4; ++l) {
    L[l].Wq = F(idx++); L[l].bq = F(idx++); L[l].Wk = F(idx++); L[l].bk = F(idx++);
    L[l].Wv = F(idx++); L[l].bv = F(idx++); L[l].Wo = F(idx++); L[l].bo = F(idx++);
    L[l].W1 = F(idx++); L[l].b1 = F(idx++); L[l].W2 = F(idx++); L[l].b2 = F(idx++);
    L[l].n1g = F(idx++); L[l].n1b = F(idx++); L[l].n2g = F(idx++); L[l].n2b = F(idx++);
  }
  struct Gated { const float *W1,*b1,*W2,*b2,*Wg,*bg,*g,*be; } G[2];
  for (int gi = 0; gi < 2; ++gi) {
    G[gi].W1 = F(idx++); G[gi].b1 = F(idx++); G[gi].W2 = F(idx++); G[gi].b2 = F(idx++);
    G[gi].Wg = F(idx++); G[gi].bg = F(idx++); G[gi].g  = F(idx++); G[gi].be = F(idx++);
  }
  const float* Wt1 = F(idx++); const float* bt1 = F(idx++);
  const float* Wt2 = F(idx++); const float* bt2 = F(idx++);
  const float* Wa1 = F(idx++); const float* ba1 = F(idx++);
  const float* Wa2 = F(idx++); const float* ba2 = F(idx++);
  const float* Wc  = F(idx++); const float* bc  = F(idx++);

  // ---- workspace layout ----
  const size_t XB = (size_t)MT * Hdim * sizeof(float);         // 235 MB
  const size_t AB = (size_t)MT * Hdim * sizeof(__bf16);        // 117 MB
  const size_t UB = 3 * XB;                                    // 705 MB union
  char* ws = (char*)d_ws;
  float*  x   = (float*)ws;
  __bf16* ab  = (__bf16*)(ws + XB);
  char*   u   = ws + XB + AB;
  // union region views
  float*  qb   = (float*)u;
  float*  kb   = (float*)(u + XB);
  float*  vb   = (float*)(u + 2 * XB);
  __bf16* hid  = (__bf16*)u;                                   // MLP hidden bf16
  float*  xm   = (float*)u;                                    // [8192,1024]
  float*  hbuf = (float*)(u + ((size_t)64  << 20));
  float*  gbuf = (float*)(u + ((size_t)128 << 20));
  __bf16* h1b  = (__bf16*)(u + ((size_t)192 << 20));
  __bf16* t1b  = (__bf16*)(u + ((size_t)256 << 20));
  __bf16* a1b  = (__bf16*)(u + ((size_t)288 << 20));
  __bf16* wptr = (__bf16*)(u + UB);

  // ---- convert + transpose all GEMM weights to bf16 ----
  auto cvt = [&](const float* W, int K, int N) -> __bf16* {
    __bf16* dst = wptr; wptr += (size_t)K * N;
    size_t total = (size_t)K * N;
    convert_wt<<<dim3((unsigned)((total + 255) / 256)), dim3(256), 0, stream>>>(W, dst, K, N);
    return dst;
  };
  __bf16 *WqT[4], *WkT[4], *WvT[4], *WoT[4], *W1T[4], *W2T[4];
  for (int l = 0; l < 4; ++l) {
    WqT[l] = cvt(L[l].Wq, Hdim, Hdim);
    WkT[l] = cvt(L[l].Wk, Hdim, Hdim);
    WvT[l] = cvt(L[l].Wv, Hdim, Hdim);
    WoT[l] = cvt(L[l].Wo, Hdim, Hdim);
    W1T[l] = cvt(L[l].W1, Hdim, FFN);
    W2T[l] = cvt(L[l].W2, FFN, Hdim);
  }
  __bf16 *GW1T[2], *GW2T[2], *GWgT[2];
  for (int gi = 0; gi < 2; ++gi) {
    GW1T[gi] = cvt(G[gi].W1, Hdim, Hdim);
    GW2T[gi] = cvt(G[gi].W2, Hdim, Hdim);
    GWgT[gi] = cvt(G[gi].Wg, Hdim, Hdim);
  }
  __bf16* Wt1T = cvt(Wt1, Hdim, 512);
  __bf16* Wa1T = cvt(Wa1, Hdim, 512);

  // ---- entity encoder -> x [B,7,H] ----
  const int entoff[7] = {0, 4, 6, 10, 14, 16, 20};
  const int entdim[7] = {4, 2, 4, 4, 2, 4, 2};
  for (int e = 0; e < 7; ++e)
    entity_encoder<<<Bsz, 256, 0, stream>>>(state, entoff[e], entdim[e], e,
        entW[e], entB[e], entG[e], entBe[e], type_embed + (size_t)e * Hdim, x);

  // ---- transformer stack ----
  for (int l = 0; l < 4; ++l) {
    ln_rows<<<MT, 256, 0, stream>>>(x, L[l].n1g, L[l].n1b, ab);
    launch_gemm(EPI_F32, ab, WqT[l], L[l].bq, nullptr, qb, nullptr, MT, Hdim, Hdim, stream);
    launch_gemm(EPI_F32, ab, WkT[l], L[l].bk, nullptr, kb, nullptr, MT, Hdim, Hdim, stream);
    launch_gemm(EPI_F32, ab, WvT[l], L[l].bv, nullptr, vb, nullptr, MT, Hdim, Hdim, stream);
    attn_rope<<<Bsz * NHEADS, 64, 0, stream>>>(qb, kb, vb, ab);
    launch_gemm(EPI_RES, ab, WoT[l], L[l].bo, x, x, nullptr, MT, Hdim, Hdim, stream);
    ln_rows<<<MT, 256, 0, stream>>>(x, L[l].n2g, L[l].n2b, ab);
    launch_gemm(EPI_GELU_BF16, ab, W1T[l], L[l].b1, nullptr, nullptr, hid, MT, FFN, Hdim, stream);
    launch_gemm(EPI_RES, hid, W2T[l], L[l].b2, x, x, nullptr, MT, Hdim, FFN, stream);
  }

  // ---- mean over entities ----
  mean_t<<<(Bsz * Hdim) / 256, 256, 0, stream>>>(x, xm, ab);

  // ---- gated blocks ----
  for (int gi = 0; gi < 2; ++gi) {
    launch_gemm(EPI_GELU_BF16, ab, GW1T[gi], G[gi].b1, nullptr, nullptr, h1b, Bsz, Hdim, Hdim, stream);
    launch_gemm(EPI_F32,       h1b, GW2T[gi], G[gi].b2, nullptr, hbuf, nullptr, Bsz, Hdim, Hdim, stream);
    launch_gemm(EPI_F32,       ab,  GWgT[gi], G[gi].bg, nullptr, gbuf, nullptr, Bsz, Hdim, Hdim, stream);
    gated_combine<<<Bsz, 256, 0, stream>>>(xm, gbuf, hbuf, G[gi].g, G[gi].be, ab);
  }

  // ---- action head ----
  launch_gemm(EPI_GELU_BF16, ab, Wt1T, bt1, nullptr, nullptr, t1b, Bsz, 512, Hdim, stream);
  launch_gemm(EPI_GELU_BF16, ab, Wa1T, ba1, nullptr, nullptr, a1b, Bsz, 512, Hdim, stream);
  head_final<<<Bsz / 256, 256, 0, stream>>>(t1b, a1b, Wt2, bt2, Wa2, ba2, Wc, bc,
                                            (float*)d_out);
}
